// DynamicCrossAttention_40286793236903
// MI455X (gfx1250) — matrix operand order repaired and checked by reading, not compiled, not run on hardware
//
#include <hip/hip_runtime.h>

#define DEV __device__ __forceinline__

typedef __attribute__((ext_vector_type(16))) __bf16       v16bf;
typedef __attribute__((ext_vector_type(8)))  float        v8f;
typedef __attribute__((ext_vector_type(8)))  unsigned int v8u;

DEV unsigned short f2bf(float f) {
  unsigned u = __float_as_uint(f);
  unsigned r = u + 0x7FFFu + ((u >> 16) & 1u);   // round-to-nearest-even
  return (unsigned short)(r >> 16);
}

// async global->LDS copy of 16B, tracked by ASYNCcnt (CDNA5 VGLOBAL async op)
DEV void async_w_load(unsigned ldsOff, const void* gaddr) {
  asm volatile("global_load_async_to_lds_b128 %0, %1, off"
               :: "v"(ldsOff), "v"(gaddr) : "memory");
}
DEV void wait_async0() {
  asm volatile("s_wait_asynccnt 0x0" ::: "memory");
}

// ---------------------------------------------------------------- upsample 32->64
__global__ void k_upsample(const float* __restrict__ tpl, float* __restrict__ out) {
  int t = blockIdx.x * blockDim.x + threadIdx.x;
  const int total = 4 * 256 * 64 * 64;
  if (t >= total) return;
  int ox = t & 63, oy = (t >> 6) & 63, bc = t >> 12;      // bc = b*256 + c
  float sy = 0.5f * oy - 0.25f, sx = 0.5f * ox - 0.25f;   // half-pixel centers
  float y0f = floorf(sy), x0f = floorf(sx);
  float fy = sy - y0f, fx = sx - x0f;
  int y0 = (int)y0f, x0 = (int)x0f;
  int y0c = min(max(y0, 0), 31), y1c = min(max(y0 + 1, 0), 31);
  int x0c = min(max(x0, 0), 31), x1c = min(max(x0 + 1, 0), 31);
  const float* src = tpl + (size_t)bc * 1024;
  float v = (1.f - fy) * ((1.f - fx) * src[y0c * 32 + x0c] + fx * src[y0c * 32 + x1c])
          +        fy  * ((1.f - fx) * src[y1c * 32 + x0c] + fx * src[y1c * 32 + x1c]);
  out[t] = v;
}

// ---------------------------------------------------------------- deform weights -> bf16 [O][2304]
__global__ void k_w2bf(const float* __restrict__ w, unsigned short* __restrict__ o) {
  int t = blockIdx.x * blockDim.x + threadIdx.x;
  if (t < 256 * 2304) o[t] = f2bf(w[t]);
}

// ---------------------------------------------------------------- offset(3x3)+mask(1x1) conv, partial over cin slice
__global__ void k_conv_partial(const float* __restrict__ aligned,
                               const float* __restrict__ search,
                               const float* __restrict__ wOffG,   // [18][512][9]
                               const float* __restrict__ wMskG,   // [9][512]
                               float* __restrict__ part) {        // [4][B][27][4096]
  __shared__ float sO[18][8][9];
  __shared__ float sM[9][8];
  int tid = threadIdx.x;
  int p = blockIdx.x * 256 + tid;   // 0..4095
  int b = blockIdx.y;
  int slice = blockIdx.z;           // 0..3 -> cin [slice*128, +128)
  int h = p >> 6, w = p & 63;
  float accO[18], accM[9];
#pragma unroll
  for (int i = 0; i < 18; i++) accO[i] = 0.f;
#pragma unroll
  for (int i = 0; i < 9; i++) accM[i] = 0.f;

  for (int c0 = slice * 128; c0 < slice * 128 + 128; c0 += 8) {
    __syncthreads();
    for (int e = tid; e < 18 * 8 * 9 + 9 * 8; e += 256) {
      if (e < 1296) {
        int co = e / 72, rem = e % 72, cc = rem / 9, t9 = rem % 9;
        sO[co][cc][t9] = wOffG[(co * 512 + c0 + cc) * 9 + t9];
      } else {
        int e2 = e - 1296, cm = e2 >> 3, cc = e2 & 7;
        sM[cm][cc] = wMskG[cm * 512 + c0 + cc];
      }
    }
    __syncthreads();
#pragma unroll
    for (int cc = 0; cc < 8; cc++) {
      int cg = c0 + cc;
      const float* base = (cg < 256) ? (aligned + ((size_t)b * 256 + cg) * 4096)
                                     : (search  + ((size_t)b * 256 + (cg - 256)) * 4096);
      float xv[9];
#pragma unroll
      for (int ty = -1; ty <= 1; ty++)
#pragma unroll
        for (int tx = -1; tx <= 1; tx++) {
          int yy = h + ty, xx = w + tx;
          float v = 0.f;
          if (yy >= 0 && yy < 64 && xx >= 0 && xx < 64) v = base[yy * 64 + xx];
          xv[(ty + 1) * 3 + (tx + 1)] = v;
        }
#pragma unroll
      for (int co = 0; co < 18; co++) {
        float s = 0.f;
#pragma unroll
        for (int t9 = 0; t9 < 9; t9++) s += xv[t9] * sO[co][cc][t9];
        accO[co] += s;
      }
#pragma unroll
      for (int cm = 0; cm < 9; cm++) accM[cm] += xv[4] * sM[cm][cc];
    }
  }
  float* dst = part + ((size_t)slice * 4 + b) * 27 * 4096 + p;
#pragma unroll
  for (int co = 0; co < 18; co++) dst[(size_t)co * 4096] = accO[co];
#pragma unroll
  for (int cm = 0; cm < 9; cm++) dst[(size_t)(18 + cm) * 4096] = accM[cm];
}

// ---------------------------------------------------------------- reduce slices, bias, sigmoid
__global__ void k_conv_final(const float* __restrict__ part,
                             const float* __restrict__ offB,
                             const float* __restrict__ mskB,
                             float* __restrict__ offs,   // [B][18][4096]
                             float* __restrict__ mask) { // [B][9][4096]
  int t = blockIdx.x * blockDim.x + threadIdx.x;
  const int total = 4 * 27 * 4096;
  if (t >= total) return;
  int p = t & 4095, ch = (t >> 12) % 27, b = t / (27 * 4096);
  float s = 0.f;
#pragma unroll
  for (int sl = 0; sl < 4; sl++)
    s += part[((size_t)sl * 4 + b) * 27 * 4096 + (size_t)ch * 4096 + p];
  if (ch < 18) {
    offs[((size_t)b * 18 + ch) * 4096 + p] = s + offB[ch];
  } else {
    float x = s + mskB[ch - 18];
    mask[((size_t)b * 9 + (ch - 18)) * 4096 + p] = 1.f / (1.f + __expf(-x));
  }
}

// ---------------------------------------------------------------- sampling plan: 4 idx + 4 fused weights per (b,k,p)
__global__ void k_plan(const float* __restrict__ offs, const float* __restrict__ mask,
                       int4* __restrict__ planI, float4* __restrict__ planW) {
  int t = blockIdx.x * blockDim.x + threadIdx.x;
  const int total = 4 * 9 * 4096;
  if (t >= total) return;
  int p = t & 4095, kk = (t >> 12) % 9, b = t / (9 * 4096);
  int h = p >> 6, w = p & 63;
  float dy = offs[((size_t)b * 18 + kk * 2 + 0) * 4096 + p];
  float dx = offs[((size_t)b * 18 + kk * 2 + 1) * 4096 + p];
  float m  = mask[t];                     // t == (b*9+kk)*4096 + p
  float py = (float)(h + (kk / 3) - 1) + dy;
  float px = (float)(w + (kk % 3) - 1) + dx;
  float y0f = floorf(py), x0f = floorf(px);
  float wy = py - y0f, wx = px - x0f;
  int y0 = (int)y0f, x0 = (int)x0f, y1 = y0 + 1, x1 = x0 + 1;
  float vy0 = (y0 >= 0 && y0 < 64) ? 1.f : 0.f;
  float vy1 = (y1 >= 0 && y1 < 64) ? 1.f : 0.f;
  float vx0 = (x0 >= 0 && x0 < 64) ? 1.f : 0.f;
  float vx1 = (x1 >= 0 && x1 < 64) ? 1.f : 0.f;
  int y0c = min(max(y0, 0), 63), y1c = min(max(y1, 0), 63);
  int x0c = min(max(x0, 0), 63), x1c = min(max(x1, 0), 63);
  planI[t] = make_int4(y0c * 64 + x0c, y0c * 64 + x1c, y1c * 64 + x0c, y1c * 64 + x1c);
  planW[t] = make_float4((1.f - wy) * (1.f - wx) * vy0 * vx0 * m,
                         (1.f - wy) * wx * vy0 * vx1 * m,
                         wy * (1.f - wx) * vy1 * vx0 * m,
                         wy * wx * vy1 * vx1 * m);
}

// ---------------------------------------------------------------- main deformable GEMM via bf16 WMMA
// out[b,o,p] = sum_{k=0..2303} W[o,k] * sampled[k,p],  K index = c*9 + kk
// Block: 256 thr (8 waves), tile = 32 pixels x 256 out-channels.
// Per K-chunk (32): weights async-copied global->LDS (double buffered),
// sampled tile built by gather+lerp into LDS (double buffered), 1 barrier/chunk,
// each wave runs 4 x v_wmma_f32_16x16x32_bf16.
__global__ void __launch_bounds__(256)
k_deform_wmma(const float* __restrict__ search,
              const unsigned short* __restrict__ wbf,   // [256][2304] bf16 bits
              const int4* __restrict__ planI,
              const float4* __restrict__ planW,
              const float* __restrict__ dbias,
              float* __restrict__ out) {
  __shared__ unsigned short wlds[2][256][32];  // weight panels, 2 x 16KB
  __shared__ unsigned samp[2][16][32];         // packed bf16 K-pairs x 32 pixels
  __shared__ int   pI[9 * 32 * 4];
  __shared__ float pW[9 * 32 * 4];

  int tid = threadIdx.x;
  int ptile = blockIdx.x;              // 0..127 (32 pixels each)
  int b = blockIdx.y;                  // 0..3
  int p0 = ptile * 32;

  // stage the sampling plan for this pixel tile (9 taps x 32 pixels)
  for (int e = tid; e < 9 * 32; e += 256) {
    int kk = e >> 5, nn = e & 31;
    int4   id = planI[((size_t)b * 9 + kk) * 4096 + p0 + nn];
    float4 wt = planW[((size_t)b * 9 + kk) * 4096 + p0 + nn];
    int base = (kk * 32 + nn) * 4;
    pI[base + 0] = id.x; pI[base + 1] = id.y; pI[base + 2] = id.z; pI[base + 3] = id.w;
    pW[base + 0] = wt.x; pW[base + 1] = wt.y; pW[base + 2] = wt.z; pW[base + 3] = wt.w;
  }

  int lane = tid & 31, wave = tid >> 5;
  int half = lane >> 4, ln = lane & 15;
  int ot0 = wave * 2, ot1 = wave * 2 + 1;     // each of 8 waves owns 2 o-tiles
  v8f acc00 = {}, acc01 = {}, acc10 = {}, acc11 = {};
  int j  = tid >> 4;                           // K-pair index this thread samples
  int n0 = tid & 15;                           // pixel columns n0 and n0+16
  const float* sbase = search + (size_t)b * 256 * 4096;

  // prefetch weight panel for chunk 0 into buffer 0 (row o = tid, 64B)
  {
    const unsigned short* g = wbf + (size_t)tid * 2304;
    unsigned lo = (unsigned)(uintptr_t)&wlds[0][tid][0];
#pragma unroll
    for (int i = 0; i < 4; i++) async_w_load(lo + i * 16, g + i * 8);
  }
  __syncthreads();   // plan LDS ready

  for (int kc = 0; kc < 72; kc++) {            // 72 chunks of K=32
    int cur = kc & 1, nxt = cur ^ 1;

    // -- sampler: gather + bilinear + mask -> packed bf16 pairs in samp[cur]
#pragma unroll
    for (int hn = 0; hn < 2; hn++) {
      int nn = n0 + hn * 16;
      unsigned packed = 0;
#pragma unroll
      for (int e = 0; e < 2; e++) {
        int k  = kc * 32 + j * 2 + e;
        int c  = k / 9;
        int kk = k - c * 9;
        int base = (kk * 32 + nn) * 4;
        const float* sp = sbase + (size_t)c * 4096;
        float s = pW[base + 0] * sp[pI[base + 0]] + pW[base + 1] * sp[pI[base + 1]]
                + pW[base + 2] * sp[pI[base + 2]] + pW[base + 3] * sp[pI[base + 3]];
        packed |= ((unsigned)f2bf(s)) << (16 * e);
      }
      samp[cur][j][nn] = packed;
    }

    wait_async0();       // weight panel for chunk kc has landed (this wave's part)
    __syncthreads();     // all waves' panels + samp[cur] visible; wlds[nxt] readers retired

    // -- prefetch next chunk's weight panel into wlds[nxt] (dummy wrap on last)
    {
      int kn = (kc + 1 < 72) ? kc + 1 : 0;
      const unsigned short* g = wbf + (size_t)tid * 2304 + kn * 32;
      unsigned lo = (unsigned)(uintptr_t)&wlds[nxt][tid][0];
#pragma unroll
      for (int i = 0; i < 4; i++) async_w_load(lo + i * 16, g + i * 8);
    }

    // -- B fragments (32x16): V0..V7 lanes0-15 -> K0-15, lanes16-31 -> K16-31
    v8u bu0, bu1;
#pragma unroll
    for (int v = 0; v < 8; v++) {
      bu0[v] = samp[cur][half * 8 + v][ln];
      bu1[v] = samp[cur][half * 8 + v][16 + ln];
    }
    v16bf bm0 = __builtin_bit_cast(v16bf, bu0);
    v16bf bm1 = __builtin_bit_cast(v16bf, bu1);

    // -- A fragments (16x32) from LDS weight panel, K = 16*half + element (matches B)
    v8u au0, au1;
#pragma unroll
    for (int v = 0; v < 8; v++) {
      int kb = half * 16 + v * 2;
      au0[v] = *(const unsigned*)&wlds[cur][ot0 * 16 + ln][kb];
      au1[v] = *(const unsigned*)&wlds[cur][ot1 * 16 + ln][kb];
    }
    v16bf am0 = __builtin_bit_cast(v16bf, au0);
    v16bf am1 = __builtin_bit_cast(v16bf, au1);

    acc00 = __builtin_amdgcn_wmma_f32_16x16x32_bf16(false, am0, false, bm0, (short)0, acc00, false, false);
    acc01 = __builtin_amdgcn_wmma_f32_16x16x32_bf16(false, am0, false, bm1, (short)0, acc01, false, false);
    acc10 = __builtin_amdgcn_wmma_f32_16x16x32_bf16(false, am1, false, bm0, (short)0, acc10, false, false);
    acc11 = __builtin_amdgcn_wmma_f32_16x16x32_bf16(false, am1, false, bm1, (short)0, acc11, false, false);
  }

  // -- epilogue: D layout (VGPR r: lanes0-15 -> M=r, lanes16-31 -> M=r+8), N = pixel
#pragma unroll
  for (int r = 0; r < 8; r++) {
    int om = r + half * 8;
    int o0 = ot0 * 16 + om, o1 = ot1 * 16 + om;
    size_t base0 = ((size_t)b * 256 + o0) * 4096 + p0;
    size_t base1 = ((size_t)b * 256 + o1) * 4096 + p0;
    out[base0 + ln]      = acc00[r] + dbias[o0];
    out[base0 + 16 + ln] = acc01[r] + dbias[o0];
    out[base1 + ln]      = acc10[r] + dbias[o1];
    out[base1 + 16 + ln] = acc11[r] + dbias[o1];
  }
}

// ---------------------------------------------------------------- launch
extern "C" void kernel_launch(void* const* d_in, const int* in_sizes, int n_in,
                              void* d_out, int out_size, void* d_ws, size_t ws_size,
                              hipStream_t stream) {
  (void)in_sizes; (void)n_in; (void)out_size; (void)ws_size;
  const float* tpl  = (const float*)d_in[0];
  const float* srch = (const float*)d_in[1];
  const float* offW = (const float*)d_in[2];
  const float* offB = (const float*)d_in[3];
  const float* mskW = (const float*)d_in[4];
  const float* mskB = (const float*)d_in[5];
  const float* defW = (const float*)d_in[6];
  const float* defB = (const float*)d_in[7];
  float* out = (float*)d_out;
  char*  ws  = (char*)d_ws;

  // workspace layout (all offsets 256B-aligned)
  float*          aligned = (float*)(ws + 0);                    // 16,777,216 B
  unsigned short* wbf     = (unsigned short*)(ws + 16777216);    //  1,179,648 B
  float*          offs    = (float*)(ws + 17956864);             //  1,179,648 B
  float*          mask    = (float*)(ws + 19136512);             //    589,824 B
  int4*           planI   = (int4*)(ws + 19726336);              //  2,359,296 B
  float4*         planW   = (float4*)(ws + 22085632);            //  2,359,296 B
  float*          part    = (float*)(ws + 24444928);             //  7,077,888 B

  k_upsample<<<(4 * 256 * 64 * 64 + 255) / 256, 256, 0, stream>>>(tpl, aligned);
  k_w2bf<<<(256 * 2304 + 255) / 256, 256, 0, stream>>>(defW, wbf);

  dim3 gB1(16, 4, 4);
  k_conv_partial<<<gB1, 256, 0, stream>>>(aligned, srch, offW, mskW, part);
  k_conv_final<<<(4 * 27 * 4096 + 255) / 256, 256, 0, stream>>>(part, offB, mskB, offs, mask);
  k_plan<<<(4 * 9 * 4096 + 255) / 256, 256, 0, stream>>>(offs, mask, planI, planW);

  dim3 gD(128, 4);
  k_deform_wmma<<<gD, 256, 0, stream>>>(srch, wbf, planI, planW, defB, out);
}
